// QNet_14156212208112
// MI455X (gfx1250) — matrix operand order repaired and checked by reading, not compiled, not run on hardware
//
#include <hip/hip_runtime.h>
#include <hip/hip_bf16.h>

typedef __attribute__((ext_vector_type(16))) _Float16 v16h;
typedef __attribute__((ext_vector_type(8)))  _Float16 v8h;
typedef __attribute__((ext_vector_type(8)))  float    v8f;

#define THREADS 256
#define WAVES   8
#define MTILES  4           // 4 x 16 = 64 rows = all actions for one state
#define N_ACT   64
#define D_STATE 5
#define D_ACT   4
#define PAD     8           // f16 row padding: bank = 4*row mod 64 -> conflict-free

__device__ __forceinline__ float fast_tanh(float x) {
#if __has_builtin(__builtin_amdgcn_tanhf)
  return __builtin_amdgcn_tanhf(x);
#else
  return tanhf(x);
#endif
}

__device__ __forceinline__ v16h load_afrag(const _Float16* __restrict__ p) {
  v8h lo = *(const v8h*)(p);
  v8h hi = *(const v8h*)(p + 8);
  return __builtin_shufflevector(lo, hi, 0,1,2,3,4,5,6,7,8,9,10,11,12,13,14,15);
}

// out[64][N] = tanh(in[64][K] @ W + bias); W^T supplied as wt[N][K] f16.
// in/out in LDS with row stride K+PAD / N+PAD.
// Wide layers (NT >= 2*WAVES): each wave accumulates TWO N-tiles at once so
// every A-fragment LDS load feeds 2 WMMAs and every B load feeds 4 WMMAs.
template<int K, int N>
__device__ __forceinline__ void gemm_layer(
    const _Float16* __restrict__ wt, const float* __restrict__ bias,
    const _Float16* __restrict__ in, _Float16* __restrict__ out,
    int lane, int wave)
{
  constexpr int LDI = K + PAD;
  constexpr int LDO = N + PAD;
  constexpr int NT  = N / 16;
  constexpr int KS  = K / 32;
  const int ln = lane & 15;
  const int hf = lane >> 4;
  const _Float16* abase = in + ln * LDI + hf * 16;

  if constexpr (NT >= 2 * WAVES) {
    for (int nt = wave; nt < NT; nt += 2 * WAVES) {
      v8f acc0[MTILES], acc1[MTILES];
#pragma unroll
      for (int mt = 0; mt < MTILES; ++mt) { acc0[mt] = (v8f){}; acc1[mt] = (v8f){}; }
      const int n0 = nt * 16 + ln;
      const int n1 = (nt + WAVES) * 16 + ln;
      const _Float16* wrow0 = wt + (size_t)n0 * K + hf * 16;
      const _Float16* wrow1 = wt + (size_t)n1 * K + hf * 16;
#pragma unroll 2
      for (int ks = 0; ks < KS; ++ks) {
        v16h bf0 = *(const v16h*)(wrow0 + ks * 32);
        v16h bf1 = *(const v16h*)(wrow1 + ks * 32);
#pragma unroll
        for (int mt = 0; mt < MTILES; ++mt) {
          v16h a = load_afrag(abase + mt * 16 * LDI + ks * 32);
          acc0[mt] = __builtin_amdgcn_wmma_f32_16x16x32_f16(false, a, false, bf0, (short)0, acc0[mt], false, false);
          acc1[mt] = __builtin_amdgcn_wmma_f32_16x16x32_f16(false, a, false, bf1, (short)0, acc1[mt], false, false);
        }
      }
      float bn0 = bias[n0];
      float bn1 = bias[n1];
#pragma unroll
      for (int mt = 0; mt < MTILES; ++mt) {
#pragma unroll
        for (int v = 0; v < 8; ++v) {
          int m = mt * 16 + v + hf * 8;  // C/D layout: VGPR v holds rows v / v+8
          out[m * LDO + n0] = (_Float16)fast_tanh(acc0[mt][v] + bn0);
          out[m * LDO + n1] = (_Float16)fast_tanh(acc1[mt][v] + bn1);
        }
      }
    }
  } else {
    for (int nt = wave; nt < NT; nt += WAVES) {
      v8f acc[MTILES];
#pragma unroll
      for (int mt = 0; mt < MTILES; ++mt) acc[mt] = (v8f){};
      const int n = nt * 16 + ln;
      const _Float16* wrow = wt + (size_t)n * K + hf * 16;
#pragma unroll 2
      for (int ks = 0; ks < KS; ++ks) {
        v16h bf = *(const v16h*)(wrow + ks * 32);
#pragma unroll
        for (int mt = 0; mt < MTILES; ++mt) {
          v16h a = load_afrag(abase + mt * 16 * LDI + ks * 32);
          acc[mt] = __builtin_amdgcn_wmma_f32_16x16x32_f16(false, a, false, bf, (short)0, acc[mt], false, false);
        }
      }
      float bn = bias[n];
#pragma unroll
      for (int mt = 0; mt < MTILES; ++mt) {
#pragma unroll
        for (int v = 0; v < 8; ++v) {
          int m = mt * 16 + v + hf * 8;
          out[m * LDO + n] = (_Float16)fast_tanh(acc[mt][v] + bn);
        }
      }
    }
  }
}

__global__ __launch_bounds__(THREADS)
void qnet_fused(
    const float* __restrict__ x,        // [B,5]
    const float* __restrict__ actions,  // [64,4]
    const float* __restrict__ W1, const float* __restrict__ b1, // [9,128],[128]
    const float* __restrict__ b2,       // [1024]
    const float* __restrict__ b3,       // [1024]
    const float* __restrict__ b4,       // [128]
    const float* __restrict__ W5, const float* __restrict__ b5, // [128],[1]
    const _Float16* __restrict__ wt2,   // [1024][128]
    const _Float16* __restrict__ wt3,   // [1024][1024]
    const _Float16* __restrict__ wt4,   // [128][1024]
    float* __restrict__ out_best,       // [B,4]
    float* __restrict__ out_vals)       // [B,64]
{
  __shared__ __align__(32) _Float16 bufA[N_ACT * (1024 + PAD)];  // ~129 KB
  __shared__ __align__(32) _Float16 bufB[N_ACT * (1024 + PAD)];  // ~129 KB
  __shared__ float vals[N_ACT];

  const int b    = blockIdx.x;
  const int tid  = threadIdx.x;
  const int lane = tid & 31;
  const int wave = tid >> 5;

  float xs[D_STATE];
#pragma unroll
  for (int k = 0; k < D_STATE; ++k) xs[k] = x[b * D_STATE + k];

  // ---- Layer 1: concat(state, action) (K=9) -> 128; tiny, f32 VALU
  for (int idx = tid; idx < N_ACT * 128; idx += THREADS) {
    int r = idx >> 7;          // row == action index
    int n = idx & 127;
    float s = b1[n];
#pragma unroll
    for (int k = 0; k < D_STATE; ++k) s += xs[k] * W1[k * 128 + n];
#pragma unroll
    for (int k = 0; k < D_ACT; ++k)  s += actions[r * D_ACT + k] * W1[(D_STATE + k) * 128 + n];
    bufA[r * (128 + PAD) + n] = (_Float16)fast_tanh(s);
  }
  __syncthreads();
  // ---- Layer 2: 128 -> 1024 (WMMA, paired N-tiles)
  gemm_layer<128, 1024>(wt2, b2, bufA, bufB, lane, wave);
  __syncthreads();
  // ---- Layer 3: 1024 -> 1024 (WMMA, dominant, paired N-tiles)
  gemm_layer<1024, 1024>(wt3, b3, bufB, bufA, lane, wave);
  __syncthreads();
  // ---- Layer 4: 1024 -> 128 (WMMA, single N-tile per wave)
  gemm_layer<1024, 128>(wt4, b4, bufA, bufB, lane, wave);
  __syncthreads();
  // ---- Layer 5: 128 -> 1 dot product per row, f32 (waves 0-1)
  if (tid < N_ACT) {
    float s = b5[0];
    for (int k = 0; k < 128; ++k) s += (float)bufB[tid * (128 + PAD) + k] * W5[k];
    vals[tid] = s;
    out_vals[b * N_ACT + tid] = s;
  }
  __syncthreads();

  // ---- argmax over 64 actions (first max wins, matches jnp.argmax)
  if (tid < 32) {
    float bv = vals[tid];
    int   bi = tid;
    float v2 = vals[tid + 32];
    if (v2 > bv) { bv = v2; bi = tid + 32; }
    for (int off = 16; off > 0; off >>= 1) {
      float ov = __shfl_down(bv, off, 32);
      int   oi = __shfl_down(bi, off, 32);
      if (ov > bv || (ov == bv && oi < bi)) { bv = ov; bi = oi; }
    }
    if (tid == 0) {
#pragma unroll
      for (int c = 0; c < D_ACT; ++c) out_best[b * D_ACT + c] = actions[bi * D_ACT + c];
    }
  }
}

// Convert + transpose the three big weight matrices to f16 [N][K] in d_ws
// so WMMA B-fragments are contiguous 32-byte loads.
__global__ __launch_bounds__(256)
void prep_weights(const float* __restrict__ W2, const float* __restrict__ W3,
                  const float* __restrict__ W4,
                  _Float16* __restrict__ wt2, _Float16* __restrict__ wt3,
                  _Float16* __restrict__ wt4)
{
  int i = blockIdx.x * 256 + threadIdx.x;
  if (i < 128 * 1024) {                 // wt2[n][k] = W2[k][n], n<1024, k<128
    int n = i >> 7, k = i & 127;
    wt2[i] = (_Float16)W2[k * 1024 + n];
  }
  if (i < 1024 * 1024) {                // wt3[n][k] = W3[k][n]
    int n = i >> 10, k = i & 1023;
    wt3[i] = (_Float16)W3[k * 1024 + n];
  }
  if (i < 128 * 1024) {                 // wt4[n][k] = W4[k][n], n<128, k<1024
    int n = i >> 10, k = i & 1023;
    wt4[i] = (_Float16)W4[k * 128 + n];
  }
}

extern "C" void kernel_launch(void* const* d_in, const int* in_sizes, int n_in,
                              void* d_out, int out_size, void* d_ws, size_t ws_size,
                              hipStream_t stream) {
  const float* x  = (const float*)d_in[0];
  const float* ac = (const float*)d_in[1];
  const float* W1 = (const float*)d_in[2];
  const float* b1 = (const float*)d_in[3];
  const float* W2 = (const float*)d_in[4];
  const float* b2 = (const float*)d_in[5];
  const float* W3 = (const float*)d_in[6];
  const float* b3 = (const float*)d_in[7];
  const float* W4 = (const float*)d_in[8];
  const float* b4 = (const float*)d_in[9];
  const float* W5 = (const float*)d_in[10];
  const float* b5 = (const float*)d_in[11];

  const int B = in_sizes[0] / D_STATE;   // 2048

  _Float16* wt2 = (_Float16*)d_ws;
  _Float16* wt3 = wt2 + 1024 * 128;
  _Float16* wt4 = wt3 + 1024 * 1024;

  prep_weights<<<(1024 * 1024 + 255) / 256, 256, 0, stream>>>(W2, W3, W4, wt2, wt3, wt4);

  float* out_best = (float*)d_out;
  float* out_vals = out_best + (size_t)B * D_ACT;
  qnet_fused<<<B, THREADS, 0, stream>>>(x, ac, W1, b1, b2, b3, b4, W5, b5,
                                        wt2, wt3, wt4, out_best, out_vals);
}
